// Model_13039520711015
// MI455X (gfx1250) — matrix operand order repaired and checked by reading, not compiled, not run on hardware
//
#include <hip/hip_runtime.h>
#include <stdint.h>

typedef __bf16 bf16_t;
typedef __attribute__((ext_vector_type(8)))  __bf16 v8bf;
typedef __attribute__((ext_vector_type(16))) __bf16 v16bf;
typedef __attribute__((ext_vector_type(8)))  float  v8f;

#define ROWS   8192
#define NB     8
#define SEQ    1024
#define DMODEL 512
#define DINNER 1024
#define NHM    4
#define DHM    256
#define NHS    8
#define DHS    64
#define DFF    563
#define DFF2   1126
#define DFFPAD 568

__device__ __forceinline__ v16bf mkv16(v8bf lo, v8bf hi) {
  union { v16bf v; v8bf h[2]; } u;
  u.h[0] = lo; u.h[1] = hi;
  return u.v;
}
__device__ __forceinline__ v8f wmma_bf(v16bf a, v16bf b, v8f c) {
  return __builtin_amdgcn_wmma_f32_16x16x32_bf16(false, a, false, b, (short)0, c, false, false);
}
__device__ __forceinline__ float sigf(float x) { return 1.f / (1.f + __expf(-x)); }

// ---------------------------------------------------------------------------
// Generic bf16 GEMM: C[M,N] = A[M,K](bf16,lda) * W[K,N](bf16) (+bias) (+add)
// 256 threads = 8 waves (4 Mwaves x 2 Nwaves), BM=BN=128, BK=32.
// ---------------------------------------------------------------------------
__global__ __launch_bounds__(256) void gemm_bf16_kernel(
    const bf16_t* __restrict__ A, int lda, int K,
    const bf16_t* __restrict__ W, int N,
    const float* __restrict__ bias, const float* __restrict__ add,
    float* __restrict__ C)
{
  __shared__ bf16_t As[128][40];
  __shared__ bf16_t Ws[128][40];
  const int bm = blockIdx.y * 128;
  const int bn = blockIdx.x * 128;
  const int tid = threadIdx.x;
  const int wave = tid >> 5;
  const int lane = tid & 31;
  const int ln = lane & 15;
  const int lh = lane >> 4;
  const int mw = (wave & 3) * 32;
  const int nw = (wave >> 2) * 64;

  const v8f vz = {0.f,0.f,0.f,0.f,0.f,0.f,0.f,0.f};
  v8f acc[2][4];
  for (int i = 0; i < 2; ++i)
    for (int j = 0; j < 4; ++j) acc[i][j] = vz;

  const int ar = tid >> 1;
  const int ac = (tid & 1) * 16;

  for (int k0 = 0; k0 < K; k0 += 32) {
    // prefetch next A tile row (speculative, dropped if OOB)
    __builtin_prefetch(A + (size_t)(bm + ar) * lda + (k0 + 32 + ac), 0, 1);
    // stage A tile (row-major M x 32)
    {
      const bf16_t* src = A + (size_t)(bm + ar) * lda + (k0 + ac);
      bf16_t* dst = &As[ar][ac];
      if (k0 + ac + 16 <= K) {
        ((uint4*)dst)[0] = ((const uint4*)src)[0];
        ((uint4*)dst)[1] = ((const uint4*)src)[1];
      } else {
        for (int j = 0; j < 16; ++j)
          dst[j] = (k0 + ac + j < K) ? src[j] : (bf16_t)0.f;
      }
    }
    // stage W tile transposed into [n][k]
    for (int i = 0; i < 16; ++i) {
      int idx = tid + 256 * i;
      int kl = idx >> 7;
      int nl = idx & 127;
      int kg = k0 + kl, ng = bn + nl;
      bf16_t v = (bf16_t)0.f;
      if (kg < K && ng < N) v = W[(size_t)kg * N + ng];
      Ws[nl][kl] = v;
    }
    __syncthreads();
    v16bf af[2], bw[4];
#pragma unroll
    for (int i = 0; i < 2; ++i) {
      int m = mw + 16 * i + ln;
      v8bf lo = *(const v8bf*)&As[m][lh * 16];
      v8bf hi = *(const v8bf*)&As[m][lh * 16 + 8];
      af[i] = mkv16(lo, hi);
    }
#pragma unroll
    for (int j = 0; j < 4; ++j) {
      int n = nw + 16 * j + ln;
      bw[j] = *(const v16bf*)&Ws[n][lh * 16];
    }
#pragma unroll
    for (int i = 0; i < 2; ++i)
#pragma unroll
      for (int j = 0; j < 4; ++j)
        acc[i][j] = wmma_bf(af[i], bw[j], acc[i][j]);
    __syncthreads();
  }
#pragma unroll
  for (int i = 0; i < 2; ++i) {
#pragma unroll
    for (int j = 0; j < 4; ++j) {
      int n = bn + nw + 16 * j + ln;
      if (n < N) {
#pragma unroll
        for (int r = 0; r < 8; ++r) {
          int m = bm + mw + 16 * i + r + 8 * lh;
          float v = acc[i][j][r];
          if (bias) v += bias[n];
          if (add)  v += add[(size_t)m * N + n];
          C[(size_t)m * N + n] = v;
        }
      }
    }
  }
}

// ---------------------------------------------------------------------------
// mLSTM parallel cell, flash style. Grid (S/16, NH, B), 128 threads (4 waves).
// Exact per-row stabilizer m[q] precomputed -> no online rescaling.
// Wave 0 computes P (16x32) with WMMA qk; all waves do P@V over 64 cols each.
// ---------------------------------------------------------------------------
__global__ __launch_bounds__(128) void mlstm_flash_kernel(
    const float* __restrict__ qf, const float* __restrict__ kf,
    const float* __restrict__ vf, const float* __restrict__ cum,
    const float* __restrict__ mstar, const float* __restrict__ ip,
    float* __restrict__ H)
{
  __shared__ bf16_t Qs[16][264];
  __shared__ bf16_t Ks[32][264];
  __shared__ bf16_t VTs[256][40];
  __shared__ bf16_t Ps[16][40];
  __shared__ float cumk[32], ik[32], cq[16], mrow[16], rs[16], nrm[16];

  const int qt = blockIdx.x, h = blockIdx.y, b = blockIdx.z;
  const int qbase = qt * 16;
  const int tid = threadIdx.x;
  const int wave = tid >> 5;
  const int lane = tid & 31;
  const int ln = lane & 15;
  const int lh = lane >> 4;
  const int rbh = b * NHM + h;
  const size_t row0 = (size_t)b * SEQ;
  const int coff = h * DHM;

  for (int i = 0; i < 32; ++i) {
    int idx = tid + 128 * i;
    int r = idx >> 8, c = idx & 255;
    Qs[r][c] = (bf16_t)qf[(row0 + qbase + r) * DINNER + coff + c];
  }
  if (tid < 16) {
    int q = qbase + tid;
    float cv = cum[(size_t)rbh * (SEQ + 1) + q + 1];
    cq[tid] = cv;
    mrow[tid] = cv + mstar[(size_t)rbh * SEQ + q];
    rs[tid] = 0.f;
  }
  const v8f vz = {0.f,0.f,0.f,0.f,0.f,0.f,0.f,0.f};
  v8f oacc[4];
  for (int j = 0; j < 4; ++j) oacc[j] = vz;
  __syncthreads();

  const int nkt = (qbase + 15) / 32 + 1;
  for (int kt = 0; kt < nkt; ++kt) {
    const int kb = kt * 32;
    for (int i = 0; i < 64; ++i) {
      int idx = tid + 128 * i;
      int r = idx >> 8, c = idx & 255;
      size_t g = (row0 + kb + r) * DINNER + coff + c;
      Ks[r][c] = (bf16_t)kf[g];
      VTs[c][r] = (bf16_t)vf[g];
    }
    if (tid < 32) {
      cumk[tid] = cum[(size_t)rbh * (SEQ + 1) + kb + tid + 1];
      ik[tid]   = ip[(size_t)rbh * SEQ + kb + tid];
    }
    __syncthreads();
    if (wave == 0) {
#pragma unroll
      for (int sub = 0; sub < 2; ++sub) {
        v8f s = vz;
#pragma unroll
        for (int c8 = 0; c8 < 8; ++c8) {
          v8bf lo = *(const v8bf*)&Qs[ln][c8 * 32 + lh * 16];
          v8bf hi = *(const v8bf*)&Qs[ln][c8 * 32 + lh * 16 + 8];
          v16bf aq = mkv16(lo, hi);
          v16bf bk = *(const v16bf*)&Ks[sub * 16 + ln][c8 * 32 + lh * 16];
          s = wmma_bf(aq, bk, s);
        }
        int keyl = sub * 16 + ln;
        int key = kb + keyl;
        float ck = cumk[keyl];
        float iv = ik[keyl];
#pragma unroll
        for (int r = 0; r < 8; ++r) {
          int m = r + 8 * lh;
          float p = 0.f;
          if (key <= qbase + m)
            p = s[r] * 0.0625f * __expf(cq[m] - ck + iv - mrow[m]);
          Ps[m][keyl] = (bf16_t)p;
          float t = p;
          t += __shfl_xor(t, 1, 32);
          t += __shfl_xor(t, 2, 32);
          t += __shfl_xor(t, 4, 32);
          t += __shfl_xor(t, 8, 32);
          if (ln == 0) rs[m] += t;   // only wave 0, lanes 0 & 16 -> distinct m
        }
      }
    }
    __syncthreads();
    {
      v8bf plo = *(const v8bf*)&Ps[ln][lh * 16];
      v8bf phi = *(const v8bf*)&Ps[ln][lh * 16 + 8];
      v16bf pa = mkv16(plo, phi);
#pragma unroll
      for (int j = 0; j < 4; ++j) {
        int dcol = wave * 64 + j * 16 + ln;
        v16bf bv = *(const v16bf*)&VTs[dcol][lh * 16];
        oacc[j] = wmma_bf(pa, bv, oacc[j]);
      }
    }
    __syncthreads();
  }
  if (tid < 16) nrm[tid] = fmaxf(fabsf(rs[tid]), __expf(-mrow[tid])) + 1e-6f;
  __syncthreads();
#pragma unroll
  for (int j = 0; j < 4; ++j) {
    int col = coff + wave * 64 + j * 16 + ln;
#pragma unroll
    for (int r = 0; r < 8; ++r) {
      int m = r + 8 * lh;
      H[(row0 + qbase + m) * DINNER + col] = oacc[j][r] / nrm[m];
    }
  }
}

// ---------------------------------------------------------------------------
// Elementwise / small kernels
// ---------------------------------------------------------------------------
__global__ void cvt1d_kernel(const float* __restrict__ in, bf16_t* __restrict__ out, int n) {
  int g = blockIdx.x * 256 + threadIdx.x;
  if (g < n) out[g] = (bf16_t)in[g];
}

__global__ void cvt2d_kernel(const float* __restrict__ in, int istride,
                             bf16_t* __restrict__ out, int ostride, int cols) {
  size_t g = (size_t)blockIdx.x * 256 + threadIdx.x;
  if (g >= (size_t)ROWS * cols) return;
  int c = (int)(g % cols);
  size_t r = g / cols;
  out[r * ostride + c] = (bf16_t)in[r * istride + c];
}

__global__ __launch_bounds__(256) void embed_kernel(
    const float* __restrict__ xe, const float* __restrict__ Win,
    const float* __restrict__ bin, float* __restrict__ X) {
  size_t g = (size_t)blockIdx.x * 256 + threadIdx.x;
  if (g >= (size_t)ROWS * DMODEL) return;
  int d = (int)(g & (DMODEL - 1));
  size_t row = g >> 9;
  float a = bin[d];
#pragma unroll
  for (int c = 0; c < 7; ++c) a += xe[row * 7 + c] * Win[c * DMODEL + d];
  X[g] = a;
}

__global__ __launch_bounds__(256) void ln_kernel(
    const float* __restrict__ x, const float* __restrict__ w,
    float* __restrict__ of, bf16_t* __restrict__ oh, int C) {
  int row = blockIdx.x, t = threadIdx.x;
  const float* xr = x + (size_t)row * C;
  float s = 0.f, s2 = 0.f;
  for (int c = t; c < C; c += 256) { float v = xr[c]; s += v; s2 += v * v; }
  __shared__ float sb[256], sb2[256];
  sb[t] = s; sb2[t] = s2; __syncthreads();
  for (int o = 128; o > 0; o >>= 1) {
    if (t < o) { sb[t] += sb[t + o]; sb2[t] += sb2[t + o]; }
    __syncthreads();
  }
  float mu = sb[0] / C;
  float var = sb2[0] / C - mu * mu;
  float rstd = rsqrtf(var + 1e-5f);
  for (int c = t; c < C; c += 256) {
    float v = (xr[c] - mu) * rstd * w[c];
    if (of) of[(size_t)row * C + c] = v;
    if (oh) oh[(size_t)row * C + c] = (bf16_t)v;
  }
}

// depthwise causal conv + bias + silu
__global__ __launch_bounds__(256) void conv_kernel(
    const float* __restrict__ x, int istride, const float* __restrict__ w,
    const float* __restrict__ bias, float* __restrict__ of,
    bf16_t* __restrict__ oh, int C, int KK) {
  size_t g = (size_t)blockIdx.x * 256 + threadIdx.x;
  if (g >= (size_t)ROWS * C) return;
  int c = (int)(g % C);
  size_t row = g / C;
  int b = (int)(row >> 10);
  int s = (int)(row & 1023);
  float acc = bias[c];
  for (int j = 0; j < KK; ++j) {
    int sp = s - (KK - 1) + j;
    if (sp >= 0) acc += w[c * KK + j] * x[((size_t)b * SEQ + sp) * istride + c];
  }
  float y = acc * sigf(acc);
  if (of) of[row * (size_t)C + c] = y;
  if (oh) oh[row * (size_t)C + c] = (bf16_t)y;
}

// mLSTM i/f gates: tiny-N projection of concat(qf,kf,vf) (3072 -> 4+4)
__global__ __launch_bounds__(256) void mgates_kernel(
    const float* __restrict__ qf, const float* __restrict__ kf,
    const float* __restrict__ vf, const float* __restrict__ Wi,
    const float* __restrict__ bi, const float* __restrict__ Wf,
    const float* __restrict__ bfv, float* __restrict__ ip, float* __restrict__ fp) {
  int row = blockIdx.x, t = threadIdx.x;
  float a[8] = {0.f,0.f,0.f,0.f,0.f,0.f,0.f,0.f};
  for (int e = t; e < 3 * DINNER; e += 256) {
    float v;
    if (e < DINNER)           v = qf[(size_t)row * DINNER + e];
    else if (e < 2 * DINNER)  v = kf[(size_t)row * DINNER + e - DINNER];
    else                      v = vf[(size_t)row * DINNER + e - 2 * DINNER];
    const float* wi = Wi + (size_t)e * 4;
    const float* wf = Wf + (size_t)e * 4;
    a[0] += v * wi[0]; a[1] += v * wi[1]; a[2] += v * wi[2]; a[3] += v * wi[3];
    a[4] += v * wf[0]; a[5] += v * wf[1]; a[6] += v * wf[2]; a[7] += v * wf[3];
  }
  __shared__ float red[8][256];
  for (int o = 0; o < 8; ++o) red[o][t] = a[o];
  __syncthreads();
  for (int s = 128; s > 0; s >>= 1) {
    if (t < s) for (int o = 0; o < 8; ++o) red[o][t] += red[o][t + s];
    __syncthreads();
  }
  int b = row >> 10, si = row & 1023;
  if (t < 4)      ip[((size_t)(b * NHM + t)) * SEQ + si] = red[t][0] + bi[t];
  else if (t < 8) fp[((size_t)(b * NHM + (t - 4))) * SEQ + si] = red[t][0] + bfv[t - 4];
}

// per-(b,h) log-sigmoid cumsum + exact running stabilizer max
__global__ void mcum_kernel(const float* __restrict__ ip, const float* __restrict__ fp,
                            float* __restrict__ cum, float* __restrict__ mstar) {
  int t = threadIdx.x;
  if (t >= NB * NHM) return;
  const float* fr = fp + (size_t)t * SEQ;
  const float* ir = ip + (size_t)t * SEQ;
  float c = 0.f, ms = -3.0e38f;
  cum[(size_t)t * (SEQ + 1)] = 0.f;
  for (int s = 0; s < SEQ; ++s) {
    float f = fr[s];
    float lf = fminf(f, 0.f) - log1pf(__expf(-fabsf(f)));
    c += lf;
    cum[(size_t)t * (SEQ + 1) + s + 1] = c;
    ms = fmaxf(ms, ir[s] - c);
    mstar[(size_t)t * SEQ + s] = ms;
  }
}

// (headnorm(H)+skip*conv)*silu(z) -> bf16
__global__ __launch_bounds__(256) void mheadnorm_kernel(
    const float* __restrict__ H, const float* __restrict__ conv,
    const float* __restrict__ UP, const float* __restrict__ nw,
    const float* __restrict__ skip, bf16_t* __restrict__ out) {
  int row = blockIdx.x, t = threadIdx.x;
  int head = t >> 6, li = t & 63;
  float v[4]; float s = 0.f, s2 = 0.f;
#pragma unroll
  for (int j = 0; j < 4; ++j) {
    int c = head * DHM + li + 64 * j;
    v[j] = H[(size_t)row * DINNER + c];
    s += v[j]; s2 += v[j] * v[j];
  }
  __shared__ float sb[256], sb2[256];
  sb[t] = s; sb2[t] = s2; __syncthreads();
  for (int o = 32; o > 0; o >>= 1) {
    if (li < o) { sb[t] += sb[t + o]; sb2[t] += sb2[t + o]; }
    __syncthreads();
  }
  float mu = sb[head * 64] / 256.f;
  float var = sb2[head * 64] / 256.f - mu * mu;
  float rstd = rsqrtf(var + 1e-5f);
#pragma unroll
  for (int j = 0; j < 4; ++j) {
    int c = head * DHM + li + 64 * j;
    float hn = (v[j] - mu) * rstd * nw[c] + skip[c] * conv[(size_t)row * DINNER + c];
    float z = UP[(size_t)row * 2048 + DINNER + c];
    out[(size_t)row * DINNER + c] = (bf16_t)(hn * (z * sigf(z)));
  }
}

// sLSTM recurrent scan: one WG per (b,h), 64 threads (one per channel)
__global__ __launch_bounds__(64) void sscan_kernel(
    const float* __restrict__ gi, const float* __restrict__ gf,
    const float* __restrict__ gz, const float* __restrict__ go,
    const float* __restrict__ Ri, const float* __restrict__ Rf,
    const float* __restrict__ Rz, const float* __restrict__ Ro,
    float* __restrict__ HS) {
  int blk = blockIdx.x;
  int b = blk >> 3, h = blk & 7;
  int d = threadIdx.x;
  __shared__ bf16_t R[4][64][64];
  __shared__ float Hs[64];
  const float* Rp[4] = {Ri, Rf, Rz, Ro};
  for (int g = 0; g < 4; ++g)
    for (int i = d; i < 4096; i += 64)
      R[g][i >> 6][i & 63] = (bf16_t)Rp[g][(size_t)h * 4096 + i];
  Hs[d] = 0.f;
  float c = 0.f, n = 0.f, m = 0.f;
  __syncthreads();
  size_t base = (size_t)b * SEQ * DMODEL + h * DHS + d;
  for (int t = 0; t < SEQ; ++t) {
    float ri = 0.f, rf = 0.f, rz = 0.f, ro = 0.f;
#pragma unroll 4
    for (int e = 0; e < 64; ++e) {
      float hp = Hs[e];
      ri += hp * (float)R[0][e][d];
      rf += hp * (float)R[1][e][d];
      rz += hp * (float)R[2][e][d];
      ro += hp * (float)R[3][e][d];
    }
    size_t off = base + (size_t)t * DMODEL;
    float it = gi[off] + ri, ft = gf[off] + rf;
    float zt = gz[off] + rz, ot = go[off] + ro;
    float mn = fmaxf(ft + m, it);
    float ig = __expf(it - mn), fg = __expf(ft + m - mn);
    c = fg * c + ig * tanhf(zt);
    n = fg * n + ig;
    m = mn;
    float hv = sigf(ot) * c / fmaxf(n, 1e-6f);
    __syncthreads();
    Hs[d] = hv;
    HS[off] = hv;
    __syncthreads();
  }
}

// headnorm(HS)*w + residual -> X3
__global__ __launch_bounds__(256) void sheadnorm_kernel(
    const float* __restrict__ HS, const float* __restrict__ X2,
    const float* __restrict__ w, float* __restrict__ X3) {
  int row = blockIdx.x, t = threadIdx.x;
  int head = t >> 5, li = t & 31;
  float v[2]; float s = 0.f, s2 = 0.f;
#pragma unroll
  for (int j = 0; j < 2; ++j) {
    int c = head * DHS + li + 32 * j;
    v[j] = HS[(size_t)row * DMODEL + c];
    s += v[j]; s2 += v[j] * v[j];
  }
  __shared__ float sb[256], sb2[256];
  sb[t] = s; sb2[t] = s2; __syncthreads();
  for (int o = 16; o > 0; o >>= 1) {
    if (li < o) { sb[t] += sb[t + o]; sb2[t] += sb2[t + o]; }
    __syncthreads();
  }
  float mu = sb[head * 32] / 64.f;
  float var = sb2[head * 32] / 64.f - mu * mu;
  float rstd = rsqrtf(var + 1e-5f);
#pragma unroll
  for (int j = 0; j < 2; ++j) {
    int c = head * DHS + li + 32 * j;
    X3[(size_t)row * DMODEL + c] = X2[(size_t)row * DMODEL + c] + (v[j] - mu) * rstd * w[c];
  }
}

// gelu(a)*g with zero-padded bf16 output (stride DFFPAD)
__global__ void gelugate_kernel(const float* __restrict__ up, bf16_t* __restrict__ out) {
  size_t g = (size_t)blockIdx.x * 256 + threadIdx.x;
  if (g >= (size_t)ROWS * DFFPAD) return;
  int c = (int)(g % DFFPAD);
  size_t row = g / DFFPAD;
  float r = 0.f;
  if (c < DFF) {
    float a = up[row * DFF2 + c];
    float gt = up[row * DFF2 + DFF + c];
    float gl = 0.5f * a * (1.f + tanhf(0.7978845608f * (a + 0.044715f * a * a * a)));
    r = gl * gt;
  }
  out[row * DFFPAD + c] = (bf16_t)r;
}

__global__ __launch_bounds__(256) void outproj_kernel(
    const float* __restrict__ xn, const float* __restrict__ Wout,
    const float* __restrict__ bout, float* __restrict__ out) {
  int row = blockIdx.x, t = threadIdx.x;
  __shared__ float xr[DMODEL];
  for (int c = t; c < DMODEL; c += 256) xr[c] = xn[(size_t)row * DMODEL + c];
  __syncthreads();
  if (t < 7) {
    float a = bout[t];
    for (int k = 0; k < DMODEL; ++k) a += xr[k] * Wout[k * 7 + t];
    out[(size_t)row * 7 + t] = a;
  }
}

// ---------------------------------------------------------------------------
static inline void launch_gemm(const bf16_t* A, int lda, int K, const bf16_t* W,
                               int N, const float* bias, const float* add,
                               float* C, hipStream_t st) {
  dim3 grid((unsigned)((N + 127) / 128), (unsigned)(ROWS / 128));
  gemm_bf16_kernel<<<grid, 256, 0, st>>>(A, lda, K, W, N, bias, add, C);
}

extern "C" void kernel_launch(void* const* d_in, const int* in_sizes, int n_in,
                              void* d_out, int out_size, void* d_ws, size_t ws_size,
                              hipStream_t stream) {
  (void)in_sizes; (void)n_in; (void)out_size; (void)ws_size;
  const float* p_x_enc   = (const float*)d_in[0];
  const float* p_W_in    = (const float*)d_in[4];
  const float* p_b_in    = (const float*)d_in[5];
  const float* p_mlnw    = (const float*)d_in[6];
  const float* p_mWup    = (const float*)d_in[7];
  const float* p_mbup    = (const float*)d_in[8];
  const float* p_mconvw  = (const float*)d_in[9];
  const float* p_mconvb  = (const float*)d_in[10];
  const float* p_mWq     = (const float*)d_in[11];
  const float* p_mWk     = (const float*)d_in[12];
  const float* p_mWv     = (const float*)d_in[13];
  const float* p_mWi     = (const float*)d_in[14];
  const float* p_mbi     = (const float*)d_in[15];
  const float* p_mWf     = (const float*)d_in[16];
  const float* p_mbf     = (const float*)d_in[17];
  const float* p_mnormw  = (const float*)d_in[18];
  const float* p_mskip   = (const float*)d_in[19];
  const float* p_mWdown  = (const float*)d_in[20];
  const float* p_mbdown  = (const float*)d_in[21];
  const float* p_slnw    = (const float*)d_in[22];
  const float* p_sconvw  = (const float*)d_in[23];
  const float* p_sconvb  = (const float*)d_in[24];
  const float* p_sWi     = (const float*)d_in[25];
  const float* p_sWf     = (const float*)d_in[26];
  const float* p_sWz     = (const float*)d_in[27];
  const float* p_sWo     = (const float*)d_in[28];
  const float* p_sRi     = (const float*)d_in[29];
  const float* p_sRf     = (const float*)d_in[30];
  const float* p_sRz     = (const float*)d_in[31];
  const float* p_sRo     = (const float*)d_in[32];
  const float* p_sbi     = (const float*)d_in[33];
  const float* p_sbf     = (const float*)d_in[34];
  const float* p_sbz     = (const float*)d_in[35];
  const float* p_sbo     = (const float*)d_in[36];
  const float* p_snormw  = (const float*)d_in[37];
  const float* p_sffnlnw = (const float*)d_in[38];
  const float* p_Wffup   = (const float*)d_in[39];
  const float* p_bffup   = (const float*)d_in[40];
  const float* p_Wffdn   = (const float*)d_in[41];
  const float* p_bffdn   = (const float*)d_in[42];
  const float* p_postlnw = (const float*)d_in[43];
  const float* p_Wout    = (const float*)d_in[44];
  const float* p_bout    = (const float*)d_in[45];
  float* out = (float*)d_out;

  char* ws = (char*)d_ws;
  size_t cur = 0;
  auto alloc = [&](size_t bytes) -> void* {
    void* p = ws + cur;
    cur = (cur + bytes + 255) & ~(size_t)255;
    return p;
  };

  float*  X     = (float*) alloc((size_t)ROWS * DMODEL * 4);
  bf16_t* XNh   = (bf16_t*)alloc((size_t)ROWS * DMODEL * 2);
  float*  UP    = (float*) alloc((size_t)ROWS * 2048 * 4);
  bf16_t* XMh   = (bf16_t*)alloc((size_t)ROWS * DINNER * 2);
  float*  CONVf = (float*) alloc((size_t)ROWS * DINNER * 4);
  bf16_t* CONVh = (bf16_t*)alloc((size_t)ROWS * DINNER * 2);
  float*  QF    = (float*) alloc((size_t)ROWS * DINNER * 4);
  float*  KF    = (float*) alloc((size_t)ROWS * DINNER * 4);
  float*  VF    = (float*) alloc((size_t)ROWS * DINNER * 4);
  float*  Hbuf  = (float*) alloc((size_t)ROWS * DINNER * 4);
  float*  UPF   = (float*) alloc((size_t)ROWS * DFF2 * 4);
  bf16_t* ACTh  = (bf16_t*)alloc((size_t)ROWS * DFFPAD * 2);
  float*  ipb   = (float*) alloc((size_t)NB * NHM * SEQ * 4);
  float*  fpb   = (float*) alloc((size_t)NB * NHM * SEQ * 4);
  float*  cumb  = (float*) alloc((size_t)NB * NHM * (SEQ + 1) * 4);
  float*  mstb  = (float*) alloc((size_t)NB * NHM * SEQ * 4);
  bf16_t* wUp = (bf16_t*)alloc((size_t)DMODEL * 2048 * 2);
  bf16_t* wQ  = (bf16_t*)alloc((size_t)DINNER * DINNER * 2);
  bf16_t* wK  = (bf16_t*)alloc((size_t)DINNER * DINNER * 2);
  bf16_t* wV  = (bf16_t*)alloc((size_t)DINNER * DINNER * 2);
  bf16_t* wDn = (bf16_t*)alloc((size_t)DINNER * DMODEL * 2);
  bf16_t* wSi = (bf16_t*)alloc((size_t)DMODEL * DMODEL * 2);
  bf16_t* wSf = (bf16_t*)alloc((size_t)DMODEL * DMODEL * 2);
  bf16_t* wSz = (bf16_t*)alloc((size_t)DMODEL * DMODEL * 2);
  bf16_t* wSo = (bf16_t*)alloc((size_t)DMODEL * DMODEL * 2);
  bf16_t* wFu = (bf16_t*)alloc((size_t)DMODEL * DFF2 * 2);
  bf16_t* wFd = (bf16_t*)alloc((size_t)DFF * DMODEL * 2);

  // lifetime-safe aliases over freed regions
  bf16_t* Gated  = (bf16_t*)QF;                                       // after flash
  float*  X2     = KF;                                                // after flash
  float*  XN2f   = KF + (size_t)ROWS * DMODEL;
  bf16_t* XN2h   = (bf16_t*)VF;                                       // after flash
  bf16_t* CONV2h = (bf16_t*)((char*)VF + (size_t)ROWS * DMODEL * 2);
  float*  XN4f   = (float*)((char*)VF + (size_t)ROWS * DMODEL * 4);
  float*  giB    = UP;                                                // after z consumed
  float*  gfB    = UP + 1 * (size_t)ROWS * DMODEL;
  float*  gzB    = UP + 2 * (size_t)ROWS * DMODEL;
  float*  goB    = UP + 3 * (size_t)ROWS * DMODEL;
  float*  HSb    = CONVf;                                             // after skip consumed
  float*  X3     = CONVf + (size_t)ROWS * DMODEL;
  bf16_t* XN3h   = XNh;
  float*  X4     = X;

  auto cdiv = [](size_t a, size_t b) { return (unsigned)((a + b - 1) / b); };

  // weight conversions f32 -> bf16
  cvt1d_kernel<<<cdiv(DMODEL * 2048, 256), 256, 0, stream>>>(p_mWup, wUp, DMODEL * 2048);
  cvt1d_kernel<<<cdiv(DINNER * DINNER, 256), 256, 0, stream>>>(p_mWq, wQ, DINNER * DINNER);
  cvt1d_kernel<<<cdiv(DINNER * DINNER, 256), 256, 0, stream>>>(p_mWk, wK, DINNER * DINNER);
  cvt1d_kernel<<<cdiv(DINNER * DINNER, 256), 256, 0, stream>>>(p_mWv, wV, DINNER * DINNER);
  cvt1d_kernel<<<cdiv(DINNER * DMODEL, 256), 256, 0, stream>>>(p_mWdown, wDn, DINNER * DMODEL);
  cvt1d_kernel<<<cdiv(DMODEL * DMODEL, 256), 256, 0, stream>>>(p_sWi, wSi, DMODEL * DMODEL);
  cvt1d_kernel<<<cdiv(DMODEL * DMODEL, 256), 256, 0, stream>>>(p_sWf, wSf, DMODEL * DMODEL);
  cvt1d_kernel<<<cdiv(DMODEL * DMODEL, 256), 256, 0, stream>>>(p_sWz, wSz, DMODEL * DMODEL);
  cvt1d_kernel<<<cdiv(DMODEL * DMODEL, 256), 256, 0, stream>>>(p_sWo, wSo, DMODEL * DMODEL);
  cvt1d_kernel<<<cdiv(DMODEL * DFF2, 256), 256, 0, stream>>>(p_Wffup, wFu, DMODEL * DFF2);
  cvt1d_kernel<<<cdiv(DFF * DMODEL, 256), 256, 0, stream>>>(p_Wffdn, wFd, DFF * DMODEL);

  // ---- embed + mLSTM block ----
  embed_kernel<<<cdiv((size_t)ROWS * DMODEL, 256), 256, 0, stream>>>(p_x_enc, p_W_in, p_b_in, X);
  ln_kernel<<<ROWS, 256, 0, stream>>>(X, p_mlnw, nullptr, XNh, DMODEL);
  launch_gemm(XNh, DMODEL, DMODEL, wUp, 2048, p_mbup, nullptr, UP, stream);
  cvt2d_kernel<<<cdiv((size_t)ROWS * DINNER, 256), 256, 0, stream>>>(UP, 2048, XMh, DINNER, DINNER);
  conv_kernel<<<cdiv((size_t)ROWS * DINNER, 256), 256, 0, stream>>>(UP, 2048, p_mconvw, p_mconvb, CONVf, CONVh, DINNER, 7);
  launch_gemm(CONVh, DINNER, DINNER, wQ, DINNER, nullptr, nullptr, QF, stream);
  launch_gemm(CONVh, DINNER, DINNER, wK, DINNER, nullptr, nullptr, KF, stream);
  launch_gemm(XMh,   DINNER, DINNER, wV, DINNER, nullptr, nullptr, VF, stream);
  mgates_kernel<<<ROWS, 256, 0, stream>>>(QF, KF, VF, p_mWi, p_mbi, p_mWf, p_mbf, ipb, fpb);
  mcum_kernel<<<1, 32, 0, stream>>>(ipb, fpb, cumb, mstb);
  {
    dim3 g(SEQ / 16, NHM, NB);
    mlstm_flash_kernel<<<g, 128, 0, stream>>>(QF, KF, VF, cumb, mstb, ipb, Hbuf);
  }
  mheadnorm_kernel<<<ROWS, 256, 0, stream>>>(Hbuf, CONVf, UP, p_mnormw, p_mskip, Gated);
  launch_gemm(Gated, DINNER, DINNER, wDn, DMODEL, p_mbdown, X, X2, stream);

  // ---- sLSTM block ----
  ln_kernel<<<ROWS, 256, 0, stream>>>(X2, p_slnw, XN2f, XN2h, DMODEL);
  conv_kernel<<<cdiv((size_t)ROWS * DMODEL, 256), 256, 0, stream>>>(XN2f, DMODEL, p_sconvw, p_sconvb, nullptr, CONV2h, DMODEL, 4);
  launch_gemm(CONV2h, DMODEL, DMODEL, wSi, DMODEL, p_sbi, nullptr, giB, stream);
  launch_gemm(CONV2h, DMODEL, DMODEL, wSf, DMODEL, p_sbf, nullptr, gfB, stream);
  launch_gemm(XN2h,   DMODEL, DMODEL, wSz, DMODEL, p_sbz, nullptr, gzB, stream);
  launch_gemm(XN2h,   DMODEL, DMODEL, wSo, DMODEL, p_sbo, nullptr, goB, stream);
  sscan_kernel<<<NB * NHS, 64, 0, stream>>>(giB, gfB, gzB, goB, p_sRi, p_sRf, p_sRz, p_sRo, HSb);
  sheadnorm_kernel<<<ROWS, 256, 0, stream>>>(HSb, X2, p_snormw, X3);

  // ---- gated FFN ----
  ln_kernel<<<ROWS, 256, 0, stream>>>(X3, p_sffnlnw, nullptr, XN3h, DMODEL);
  launch_gemm(XN3h, DMODEL, DMODEL, wFu, DFF2, p_bffup, nullptr, UPF, stream);
  gelugate_kernel<<<cdiv((size_t)ROWS * DFFPAD, 256), 256, 0, stream>>>(UPF, ACTh);
  launch_gemm(ACTh, DFFPAD, DFF, wFd, DMODEL, p_bffdn, X3, X4, stream);

  // ---- post norm + output projection ----
  ln_kernel<<<ROWS, 256, 0, stream>>>(X4, p_postlnw, XN4f, nullptr, DMODEL);
  outproj_kernel<<<ROWS, 256, 0, stream>>>(XN4f, p_Wout, p_bout, out);
}
